// CLIPDenseBase_23106924053092
// MI455X (gfx1250) — matrix operand order repaired and checked by reading, not compiled, not run on hardware
//
#include <hip/hip_runtime.h>
#include <hip/hip_bf16.h>
#include <hip/hip_fp16.h>

typedef __attribute__((ext_vector_type(16))) _Float16 v16h;
typedef __attribute__((ext_vector_type(8)))  float    v8f;

union Frag16 {
    v16h v;
    _Float16 h[16];
    uint4 q[2];
};

union H8 {
    _Float16 h[8];
    uint4 q;
};

#define TB 256

// ---------------------------------------------------------------------------
// Generic batched GEMM:  C[z] = A[z] (f32, MxK, lda) * W[z]^T (f16, NxK, ldw)
//                        + bias + epilogue(mode)
// mode: 0 = none, 1 = QuickGELU(v*sigmoid(1.702v)), 2 = += Res
// Block tile 128x128, 8 waves (wave32), wave tile 32x64 (2x4 wmma tiles).
// ---------------------------------------------------------------------------
__global__ __launch_bounds__(256)
void gemm_wmma_f16(const float* __restrict__ A, const _Float16* __restrict__ W,
                   const float* __restrict__ bias, const float* __restrict__ Res,
                   float* __restrict__ C,
                   int M, int N, int K,
                   int lda, int ldw, int ldc,
                   long long strideA, long long strideW, long long strideC,
                   int mode)
{
    const int z = blockIdx.z;
    A += (long long)z * strideA;
    W += (long long)z * strideW;
    C += (long long)z * strideC;
    if (Res) Res += (long long)z * strideC;

    const int bm = blockIdx.y * 128;
    const int bn = blockIdx.x * 128;
    const int tid  = threadIdx.x;
    const int wave = tid >> 5;
    const int lane = tid & 31;
    const int wm = wave & 3;     // 4 waves along M -> 32 rows each
    const int wn = wave >> 2;    // 2 waves along N -> 64 cols each
    const int lm = lane & 15;    // row/col within 16x16 tile
    const int lh = lane >> 4;    // K-half selector

    const int LDSS = 40;         // padded LDS row stride (halfs)
    __shared__ _Float16 sA[128 * 40];

    // ---- staging geometry (fixed per thread) ------------------------------
    const int sr  = tid >> 1;           // LDS row this thread stages
    const int sc0 = (tid & 1) * 16;     // first of 16 staged columns
    const int gr  = bm + sr;
    const bool rowin = (gr < M);
    const int grc = rowin ? gr : (M - 1);           // clamped row
    const float* __restrict__ arow = A + (long long)grc * lda;
    const bool ldaOK = ((lda & 3) == 0);

    // ---- B-tile columns (fixed per thread) --------------------------------
    int  bcol[4];
    bool bval[4];
#pragma unroll
    for (int j = 0; j < 4; ++j) {
        bcol[j] = bn + wn * 64 + j * 16 + lm;
        bval[j] = bcol[j] < N;
    }

    v8f acc[2][4];
    const v8f z8 = {0.f, 0.f, 0.f, 0.f, 0.f, 0.f, 0.f, 0.f};
#pragma unroll
    for (int i = 0; i < 2; ++i)
#pragma unroll
        for (int j = 0; j < 4; ++j) acc[i][j] = z8;

    for (int kb = 0; kb < K; kb += 32) {
        // ---- stage A block [128 x 32] f32 -> f16 into LDS -----------------
        if (ldaOK && (kb + 32 <= K)) {
            // fast path: 4x float4 loads, branchless row-valid select
            const float* ap = arow + kb + sc0;
            float4 f0 = *(const float4*)(ap + 0);
            float4 f1 = *(const float4*)(ap + 4);
            float4 f2 = *(const float4*)(ap + 8);
            float4 f3 = *(const float4*)(ap + 12);
            if (kb + 64 <= K) __builtin_prefetch(ap + 32, 0, 1);
            const float rs = rowin ? 1.f : 0.f;
            H8 p0, p1;
            p0.h[0] = (_Float16)(f0.x * rs); p0.h[1] = (_Float16)(f0.y * rs);
            p0.h[2] = (_Float16)(f0.z * rs); p0.h[3] = (_Float16)(f0.w * rs);
            p0.h[4] = (_Float16)(f1.x * rs); p0.h[5] = (_Float16)(f1.y * rs);
            p0.h[6] = (_Float16)(f1.z * rs); p0.h[7] = (_Float16)(f1.w * rs);
            p1.h[0] = (_Float16)(f2.x * rs); p1.h[1] = (_Float16)(f2.y * rs);
            p1.h[2] = (_Float16)(f2.z * rs); p1.h[3] = (_Float16)(f2.w * rs);
            p1.h[4] = (_Float16)(f3.x * rs); p1.h[5] = (_Float16)(f3.y * rs);
            p1.h[6] = (_Float16)(f3.z * rs); p1.h[7] = (_Float16)(f3.w * rs);
            *(uint4*)&sA[sr * LDSS + sc0 + 0] = p0.q;
            *(uint4*)&sA[sr * LDSS + sc0 + 8] = p1.q;
        } else {
            // edge path: unconditional clamped loads + per-element select
            H8 p0, p1;
#pragma unroll
            for (int c = 0; c < 8; ++c) {
                const int gk = kb + sc0 + c;
                const int gkc = (gk < K) ? gk : (K - 1);
                float v = arow[gkc];
                p0.h[c] = (_Float16)((rowin && gk < K) ? v : 0.f);
            }
#pragma unroll
            for (int c = 0; c < 8; ++c) {
                const int gk = kb + sc0 + 8 + c;
                const int gkc = (gk < K) ? gk : (K - 1);
                float v = arow[gkc];
                p1.h[c] = (_Float16)((rowin && gk < K) ? v : 0.f);
            }
            *(uint4*)&sA[sr * LDSS + sc0 + 0] = p0.q;
            *(uint4*)&sA[sr * LDSS + sc0 + 8] = p1.q;
        }
        __syncthreads();

        // ---- A fragments from LDS (16x32 f16 per tile) --------------------
        Frag16 af[2];
#pragma unroll
        for (int i = 0; i < 2; ++i) {
            const int row = wm * 32 + i * 16 + lm;
            af[i].q[0] = *(const uint4*)&sA[row * LDSS + lh * 8];
            af[i].q[1] = *(const uint4*)&sA[row * LDSS + 16 + lh * 8];
        }

        // ---- B fragments straight from global f16 weights -----------------
        // lane holds column n = lm of B (= row n of W), K = lh*8..+7 and 16+lh*8..+7
        Frag16 bf[4];
#pragma unroll
        for (int j = 0; j < 4; ++j) {
            if (bval[j]) {
                const _Float16* wp = W + (long long)bcol[j] * ldw + kb + lh * 8;
                bf[j].q[0] = *(const uint4*)wp;
                bf[j].q[1] = *(const uint4*)(wp + 16);
                if (kb + 64 <= K) __builtin_prefetch(wp + 32, 0, 1);
            } else {
                bf[j].q[0] = make_uint4(0u, 0u, 0u, 0u);
                bf[j].q[1] = make_uint4(0u, 0u, 0u, 0u);
            }
        }

        // ---- 8 WMMAs per wave per K-step ----------------------------------
#pragma unroll
        for (int i = 0; i < 2; ++i)
#pragma unroll
            for (int j = 0; j < 4; ++j)
                acc[i][j] = __builtin_amdgcn_wmma_f32_16x16x32_f16(
                    false, af[i].v, false, bf[j].v,
                    (short)0, acc[i][j], false, false);

        __syncthreads();
    }

    // ---- epilogue: C/D layout m = lh*8 + r, n = lm ------------------------
#pragma unroll
    for (int i = 0; i < 2; ++i) {
        const int mb = bm + wm * 32 + i * 16 + lh * 8;
#pragma unroll
        for (int j = 0; j < 4; ++j) {
            const int gn = bcol[j];
            if (gn >= N) continue;
            const float bv = bias ? bias[gn] : 0.f;
#pragma unroll
            for (int r = 0; r < 8; ++r) {
                const int gm = mb + r;
                if (gm >= M) continue;
                float v = acc[i][j][r] + bv;
                if (mode == 1) v = v / (1.f + __expf(-1.702f * v)); // QuickGELU
                if (mode == 2) v += Res[(long long)gm * ldc + gn];
                C[(long long)gm * ldc + gn] = v;
            }
        }
    }
}

// ---------------------------------------------------------------------------
// Elementwise / helper kernels
// ---------------------------------------------------------------------------
__global__ void wcvt_kernel(const float* __restrict__ s, _Float16* __restrict__ d,
                            long long n)
{
    long long i = (long long)blockIdx.x * TB + threadIdx.x;
    if (i < n) d[i] = (_Float16)s[i];
}

// proj [768,512] f32  ->  projT [512,768] f16
__global__ void tcvt_kernel(const float* __restrict__ s, _Float16* __restrict__ d,
                            int K, int N)
{
    long long i = (long long)blockIdx.x * TB + threadIdx.x;
    if (i < (long long)K * N) {
        int k = (int)(i / N), n = (int)(i % N);
        d[(long long)n * K + k] = (_Float16)s[i];
    }
}

// x_inp [B,3,224,224] -> patches [B*196, 768]   (col = c*256 + py*16 + px)
__global__ void im2col_kernel(const float* __restrict__ x, float* __restrict__ p,
                              long long total)
{
    long long i = (long long)blockIdx.x * TB + threadIdx.x;
    if (i >= total) return;
    int col = (int)(i % 768);
    long long row = i / 768;
    int b = (int)(row / 196), t = (int)(row % 196);
    int ph = t / 14, pw = t % 14;
    int c = col / 256, rr = col % 256;
    int py = rr / 16, px = rr % 16;
    p[i] = x[(((long long)b * 3 + c) * 224 + ph * 16 + py) * 224 + pw * 16 + px];
}

// assemble [B,197,768]: cls token + patch tokens + pos_emb
__global__ void embed_kernel(const float* __restrict__ pout,
                             const float* __restrict__ cls_e,
                             const float* __restrict__ pos,
                             float* __restrict__ xa, long long total)
{
    long long i = (long long)blockIdx.x * TB + threadIdx.x;
    if (i >= total) return;
    int d = (int)(i % 768);
    int s = (int)((i / 768) % 197);
    long long b = i / (768LL * 197);
    float v = (s == 0) ? cls_e[d] : pout[(b * 196 + s - 1) * 768 + d];
    xa[i] = v + pos[s * 768 + d];
}

__global__ void ln_kernel(const float* __restrict__ in, float* __restrict__ out,
                          const float* __restrict__ w, const float* __restrict__ b,
                          int cols, long long istride, long long ostride)
{
    const long long row = blockIdx.x;
    const float* ip = in + row * istride;
    float* op = out + row * ostride;
    __shared__ float s1[TB], s2[TB];
    float sum = 0.f, sq = 0.f;
    for (int c = threadIdx.x; c < cols; c += TB) {
        float v = ip[c]; sum += v; sq += v * v;
    }
    s1[threadIdx.x] = sum; s2[threadIdx.x] = sq;
    __syncthreads();
    for (int s = TB / 2; s > 0; s >>= 1) {
        if (threadIdx.x < s) {
            s1[threadIdx.x] += s1[threadIdx.x + s];
            s2[threadIdx.x] += s2[threadIdx.x + s];
        }
        __syncthreads();
    }
    const float mean = s1[0] / cols;
    const float var  = s2[0] / cols - mean * mean;
    const float inv  = rsqrtf(var + 1e-5f);
    for (int c = threadIdx.x; c < cols; c += TB)
        op[c] = (ip[c] - mean) * inv * w[c] + b[c];
}

// qkv [B,S,2304] -> q_pack f32 [Z,S,64], k f16 [Z,S,64], v^T f16 [Z,64,224]
__global__ void pack_qkv_kernel(const float* __restrict__ qkv,
                                float* __restrict__ qp,
                                _Float16* __restrict__ kp,
                                _Float16* __restrict__ vt, long long total)
{
    long long i = (long long)blockIdx.x * TB + threadIdx.x;
    if (i >= total) return;
    int d = (int)(i % 64);
    int s = (int)((i / 64) % 197);
    int zz = (int)(i / (64LL * 197));
    int b = zz / 12, h = zz % 12;
    long long base = ((long long)b * 197 + s) * 2304 + h * 64 + d;
    qp[i] = qkv[base];
    kp[i] = (_Float16)qkv[base + 768];
    vt[(long long)zz * 64 * 224 + (long long)d * 224 + s] = (_Float16)qkv[base + 1536];
}

// row softmax of scale*s, in place; one block per row; row stride ld
__global__ void softmax_kernel(float* __restrict__ sc, int cols, int ld, float scale)
{
    float* p = sc + (long long)blockIdx.x * ld;
    __shared__ float red[TB];
    float m = -1e30f;
    for (int c = threadIdx.x; c < cols; c += TB) m = fmaxf(m, p[c]);
    red[threadIdx.x] = m;
    __syncthreads();
    for (int s = TB / 2; s > 0; s >>= 1) {
        if (threadIdx.x < s) red[threadIdx.x] = fmaxf(red[threadIdx.x], red[threadIdx.x + s]);
        __syncthreads();
    }
    m = red[0];
    __syncthreads();
    float sum = 0.f;
    for (int c = threadIdx.x; c < cols; c += TB) {
        float e = __expf((p[c] - m) * scale);
        p[c] = e; sum += e;
    }
    red[threadIdx.x] = sum;
    __syncthreads();
    for (int s = TB / 2; s > 0; s >>= 1) {
        if (threadIdx.x < s) red[threadIdx.x] += red[threadIdx.x + s];
        __syncthreads();
    }
    const float inv = 1.f / red[0];
    for (int c = threadIdx.x; c < cols; c += TB) p[c] *= inv;
}

// o_tmp [Z=B*H, S, 64] -> attn_out [B, S, 768]
__global__ void permute_o_kernel(const float* __restrict__ ot, float* __restrict__ out,
                                 long long total)
{
    long long i = (long long)blockIdx.x * TB + threadIdx.x;
    if (i >= total) return;
    int d = (int)(i % 768);
    int s = (int)((i / 768) % 197);
    long long b = i / (768LL * 197);
    int h = d / 64, dd = d % 64;
    out[i] = ot[((b * 12 + h) * 197 + s) * 64 + dd];
}

// ---------------------------------------------------------------------------
// Host orchestration
// ---------------------------------------------------------------------------
extern "C" void kernel_launch(void* const* d_in, const int* in_sizes, int n_in,
                              void* d_out, int out_size, void* d_ws, size_t ws_size,
                              hipStream_t stream)
{
    const float* x_inp    = (const float*)d_in[0];
    const float* conv_w   = (const float*)d_in[1];
    const float* cls_emb  = (const float*)d_in[2];
    const float* pos_emb  = (const float*)d_in[3];
    const float* ln_pre_w = (const float*)d_in[4];
    const float* ln_pre_b = (const float*)d_in[5];
    const float* ln1_w    = (const float*)d_in[6];
    const float* ln1_b    = (const float*)d_in[7];
    const float* qkv_w    = (const float*)d_in[8];
    const float* qkv_b    = (const float*)d_in[9];
    const float* out_w    = (const float*)d_in[10];
    const float* out_b    = (const float*)d_in[11];
    const float* ln2_w    = (const float*)d_in[12];
    const float* ln2_b    = (const float*)d_in[13];
    const float* fc1_w    = (const float*)d_in[14];
    const float* fc1_b    = (const float*)d_in[15];
    const float* fc2_w    = (const float*)d_in[16];
    const float* fc2_b    = (const float*)d_in[17];
    const float* lnp_w    = (const float*)d_in[18];
    const float* lnp_b    = (const float*)d_in[19];
    const float* proj     = (const float*)d_in[20];

    const int Lc = 12, Dm = 768, Hh = 12, DH = 64, FF = 3072;
    const int Bb = 32, Ss = 197, NP = 196, OUTN = 512;
    const int BS = Bb * Ss;                  // 6304
    const long long BP = (long long)Bb * NP; // 6272
    const int Zb = Bb * Hh;                  // 384
    const int SP = 224;                      // padded seq for v^T (K padding)
    const int SLD = 208;                     // padded score row stride (mult of 4)

    // ---- carve workspace --------------------------------------------------
    size_t off = 0;
    auto carve = [&](size_t bytes) -> void* {
        void* p = (char*)d_ws + off;
        off = (off + bytes + 255) & ~(size_t)255;
        return p;
    };
    _Float16* hConv  = (_Float16*)carve((size_t)Dm * Dm * 2);
    _Float16* hQkv   = (_Float16*)carve((size_t)Lc * 3 * Dm * Dm * 2);
    _Float16* hOut   = (_Float16*)carve((size_t)Lc * Dm * Dm * 2);
    _Float16* hFc1   = (_Float16*)carve((size_t)Lc * FF * Dm * 2);
    _Float16* hFc2   = (_Float16*)carve((size_t)Lc * Dm * FF * 2);
    _Float16* hProjT = (_Float16*)carve((size_t)OUTN * Dm * 2);
    float* patches = (float*)carve((size_t)BP * Dm * 4);
    float* pout    = (float*)carve((size_t)BP * Dm * 4);
    float* xbuf    = (float*)carve((size_t)BS * Dm * 4);
    float* ybuf    = (float*)carve((size_t)BS * Dm * 4);
    float* qkvb    = (float*)carve((size_t)BS * 3 * Dm * 4);
    float* ffbuf   = (float*)carve((size_t)BS * FF * 4);
    float* qpk     = (float*)carve((size_t)Zb * Ss * DH * 4);
    _Float16* kf16 = (_Float16*)carve((size_t)Zb * Ss * DH * 2);
    _Float16* vt   = (_Float16*)carve((size_t)Zb * DH * SP * 2);
    float* sc      = (float*)carve((size_t)Zb * Ss * SLD * 4);
    float* otmp    = (float*)carve((size_t)Zb * Ss * DH * 4);
    float* clsb    = (float*)carve((size_t)Bb * Dm * 4);

    auto blocks = [](long long n) { return dim3((unsigned)((n + TB - 1) / TB)); };

    auto gemm = [&](const float* A, const _Float16* W, const float* bias,
                    const float* Res, float* C, int M, int N, int K,
                    int lda, int ldw, int ldc,
                    long long sA_, long long sW_, long long sC_, int Z, int mode) {
        dim3 g((N + 127) / 128, (M + 127) / 128, Z);
        gemm_wmma_f16<<<g, dim3(256), 0, stream>>>(A, W, bias, Res, C, M, N, K,
                                                   lda, ldw, ldc, sA_, sW_, sC_, mode);
    };

    // ---- weight downcast (f32 -> f16) -------------------------------------
    {
        long long n;
        n = (long long)Dm * Dm;
        wcvt_kernel<<<blocks(n), TB, 0, stream>>>(conv_w, hConv, n);
        n = (long long)Lc * 3 * Dm * Dm;
        wcvt_kernel<<<blocks(n), TB, 0, stream>>>(qkv_w, hQkv, n);
        n = (long long)Lc * Dm * Dm;
        wcvt_kernel<<<blocks(n), TB, 0, stream>>>(out_w, hOut, n);
        n = (long long)Lc * FF * Dm;
        wcvt_kernel<<<blocks(n), TB, 0, stream>>>(fc1_w, hFc1, n);
        n = (long long)Lc * Dm * FF;
        wcvt_kernel<<<blocks(n), TB, 0, stream>>>(fc2_w, hFc2, n);
        n = (long long)Dm * OUTN;
        tcvt_kernel<<<blocks(n), TB, 0, stream>>>(proj, hProjT, Dm, OUTN);
    }

    // zero v^T once so the K-padding region [197,224) stays zero
    hipMemsetAsync(vt, 0, (size_t)Zb * DH * SP * 2, stream);

    // ---- patch embedding ---------------------------------------------------
    {
        long long n = BP * Dm;
        im2col_kernel<<<blocks(n), TB, 0, stream>>>(x_inp, patches, n);
        gemm(patches, hConv, nullptr, nullptr, pout,
             (int)BP, Dm, Dm, Dm, Dm, Dm, 0, 0, 0, 1, 0);
        n = (long long)BS * Dm;
        embed_kernel<<<blocks(n), TB, 0, stream>>>(pout, cls_emb, pos_emb, ybuf, n);
        ln_kernel<<<dim3(BS), TB, 0, stream>>>(ybuf, xbuf, ln_pre_w, ln_pre_b,
                                               Dm, Dm, Dm);
    }

    const float scale = 0.125f; // DH^-0.5

    // ---- transformer blocks ------------------------------------------------
    for (int l = 0; l < Lc; ++l) {
        // y = LN1(x)
        ln_kernel<<<dim3(BS), TB, 0, stream>>>(xbuf, ybuf, ln1_w + l * Dm,
                                               ln1_b + l * Dm, Dm, Dm, Dm);
        // qkv = y @ qkv_w^T + qkv_b
        gemm(ybuf, hQkv + (size_t)l * 3 * Dm * Dm, qkv_b + l * 3 * Dm, nullptr,
             qkvb, BS, 3 * Dm, Dm, Dm, Dm, 3 * Dm, 0, 0, 0, 1, 0);
        // split heads
        {
            long long n = (long long)Zb * Ss * DH;
            pack_qkv_kernel<<<blocks(n), TB, 0, stream>>>(qkvb, qpk, kf16, vt, n);
        }
        // scores = q @ k^T   (batched over Z=B*H), row stride SLD
        gemm(qpk, kf16, nullptr, nullptr, sc, Ss, Ss, DH, DH, DH, SLD,
             (long long)Ss * DH, (long long)Ss * DH, (long long)Ss * SLD, Zb, 0);
        // softmax(scale * scores)
        softmax_kernel<<<dim3((unsigned)(Zb * Ss)), TB, 0, stream>>>(sc, Ss, SLD, scale);
        // o = P @ v   (W = v^T [64, 224] zero padded)
        gemm(sc, vt, nullptr, nullptr, otmp, Ss, DH, Ss, SLD, SP, DH,
             (long long)Ss * SLD, (long long)DH * SP, (long long)Ss * DH, Zb, 0);
        // merge heads -> y
        {
            long long n = (long long)BS * Dm;
            permute_o_kernel<<<blocks(n), TB, 0, stream>>>(otmp, ybuf, n);
        }
        // x = x + y @ out_w^T + out_b
        gemm(ybuf, hOut + (size_t)l * Dm * Dm, out_b + l * Dm, xbuf, xbuf,
             BS, Dm, Dm, Dm, Dm, Dm, 0, 0, 0, 1, 2);
        // y = LN2(x)
        ln_kernel<<<dim3(BS), TB, 0, stream>>>(xbuf, ybuf, ln2_w + l * Dm,
                                               ln2_b + l * Dm, Dm, Dm, Dm);
        // ff = QuickGELU(y @ fc1^T + b)
        gemm(ybuf, hFc1 + (size_t)l * FF * Dm, fc1_b + l * FF, nullptr, ffbuf,
             BS, FF, Dm, Dm, Dm, FF, 0, 0, 0, 1, 1);
        // x = x + ff @ fc2^T + b
        gemm(ffbuf, hFc2 + (size_t)l * Dm * FF, fc2_b + l * Dm, xbuf, xbuf,
             BS, Dm, FF, FF, FF, Dm, 0, 0, 0, 1, 2);
    }

    // ---- head: LN_post on cls token, then projection -----------------------
    ln_kernel<<<dim3(Bb), TB, 0, stream>>>(xbuf, clsb, lnp_w, lnp_b,
                                           Dm, (long long)Ss * Dm, Dm);
    gemm(clsb, hProjT, nullptr, nullptr, (float*)d_out,
         Bb, OUTN, Dm, Dm, Dm, OUTN, 0, 0, 0, 1, 0);

    (void)in_sizes; (void)n_in; (void)out_size; (void)ws_size;
}
